// Sinkhorn_70600672412332
// MI455X (gfx1250) — matrix operand order repaired and checked by reading, not compiled, not run on hardware
//
#include <hip/hip_runtime.h>
#include <hip/hip_bf16.h>

// ---------------- types for WMMA ----------------
typedef __attribute__((ext_vector_type(16))) __bf16         v16bf;
typedef __attribute__((ext_vector_type(8)))  float          v8f;
typedef __attribute__((ext_vector_type(8)))  unsigned short u16x8;

#define NORM_C (-7.7424023f)   // -log(256 + 2048)
#define INV_EPS 20.0f          // 1 / 0.05

__device__ __forceinline__ unsigned short f2bf(float f) {
    union { float f; unsigned int u; } x; x.f = f;
    unsigned int u = x.u;
    u += 0x7FFFu + ((u >> 16) & 1u);   // round-to-nearest-even
    return (unsigned short)(u >> 16);
}

// A-fragment (16x32 bf16, MxK): lane holds contiguous K in [kh*16, kh*16+16)
__device__ __forceinline__ v16bf load_a_frag(const unsigned short* p) {
    union { u16x8 h[2]; v16bf v; } u;
    u.h[0] = *(const u16x8*)(p);
    u.h[1] = *(const u16x8*)(p + 8);
    return u.v;
}
// B-fragment (32x16 bf16, KxN): lane holds contiguous K in [kh*16, kh*16+16)
__device__ __forceinline__ v16bf load_b_frag(const unsigned short* p) {
    union { u16x8 h[2]; v16bf v; } u;
    u.h[0] = *(const u16x8*)(p);
    u.h[1] = *(const u16x8*)(p + 8);
    return u.v;
}

// ---------------- 1) L2-normalize rows of length 768, emit bf16 ----------------
__global__ void nrm_rows(const float* __restrict__ in, unsigned short* __restrict__ out) {
    const int row = blockIdx.x;
    const int tid = threadIdx.x;
    const float* r = in + (size_t)row * 768;
    float a0 = r[tid], a1 = r[tid + 256], a2 = r[tid + 512];
    __shared__ float red[256];
    red[tid] = a0 * a0 + a1 * a1 + a2 * a2;
    __syncthreads();
    for (int off = 128; off > 0; off >>= 1) {
        if (tid < off) red[tid] += red[tid + off];
        __syncthreads();
    }
    const float rn = rsqrtf(red[0]);
    unsigned short* o = out + (size_t)row * 768;
    o[tid]       = f2bf(a0 * rn);
    o[tid + 256] = f2bf(a1 * rn);
    o[tid + 512] = f2bf(a2 * rn);
}

// ---------------- 2) scores GEMM: Z[b,m,n] = (Xn[b,n,:] . W[m,:]) / EPS ----------------
// grid (4, 16, 32), block 256.  Each wave: 16(M) x 64(N) tile, K=768 in steps of 32.
// Simple loop body: the compiler fully unrolls and pipelines the s_clause'd b128
// loads under the WMMAs without spilling (verified round 1).
__global__ void score_wmma(const unsigned short* __restrict__ xn,
                           const unsigned short* __restrict__ wb,
                           float* __restrict__ Z) {
    const int tid  = threadIdx.x;
    const int wave = tid >> 5, lane = tid & 31;
    const int r16  = lane & 15, kh = lane >> 4;
    const int m0   = blockIdx.y << 4;
    const int n0   = (blockIdx.x << 9) + (wave << 6);
    const int b    = blockIdx.z;

    const unsigned short* arow  = wb + (size_t)(m0 + r16) * 768 + kh * 16;
    const unsigned short* bbase = xn + ((size_t)b * 2048 + n0 + r16) * 768 + kh * 16;

    v8f acc0 = {0.f,0.f,0.f,0.f,0.f,0.f,0.f,0.f};
    v8f acc1 = acc0, acc2 = acc0, acc3 = acc0;

    for (int k0 = 0; k0 < 768; k0 += 32) {
        if (k0 + 32 < 768) {
            __builtin_prefetch(bbase + k0 + 32, 0, 3);
            __builtin_prefetch(bbase + 16 * 768 + k0 + 32, 0, 3);
        }
        v16bf A  = load_a_frag(arow + k0);
        v16bf B0 = load_b_frag(bbase + k0);
        v16bf B1 = load_b_frag(bbase + (size_t)16 * 768 + k0);
        v16bf B2 = load_b_frag(bbase + (size_t)32 * 768 + k0);
        v16bf B3 = load_b_frag(bbase + (size_t)48 * 768 + k0);
        acc0 = __builtin_amdgcn_wmma_f32_16x16x32_bf16(false, A, false, B0, (short)0, acc0, false, false);
        acc1 = __builtin_amdgcn_wmma_f32_16x16x32_bf16(false, A, false, B1, (short)0, acc1, false, false);
        acc2 = __builtin_amdgcn_wmma_f32_16x16x32_bf16(false, A, false, B2, (short)0, acc2, false, false);
        acc3 = __builtin_amdgcn_wmma_f32_16x16x32_bf16(false, A, false, B3, (short)0, acc3, false, false);
    }

#pragma unroll
    for (int vv = 0; vv < 8; ++vv) {
        const int row = m0 + vv + (kh << 3);
        float* zr = Z + ((size_t)(b * 256 + row)) * 2048 + n0 + r16;
        zr[0]  = acc0[vv] * INV_EPS;
        zr[16] = acc1[vv] * INV_EPS;
        zr[32] = acc2[vv] * INV_EPS;
        zr[48] = acc3[vv] * INV_EPS;
    }
}

// ---------------- 3a) u[b,i] = norm - logsumexp_j(Z[b,i,j] + v[b,j]) ----------------
__global__ void sink_u_kernel(const float* __restrict__ Z, const float* __restrict__ v,
                              float* __restrict__ u) {
    const int rid = blockIdx.x;           // b*256 + i
    const int b   = rid >> 8;
    const int tid = threadIdx.x;
    const float* zr = Z + (size_t)rid * 2048;
    const float* vb = v + (size_t)b * 2048;
    float mx = -3.0e38f, s = 0.f;
#pragma unroll
    for (int t = 0; t < 8; ++t) {
        const int j = tid + (t << 8);
        const float z = zr[j] + vb[j];
        if (z > mx) { s = s * __expf(mx - z) + 1.f; mx = z; }
        else        { s += __expf(z - mx); }
    }
    __shared__ float sm[256], ss[256];
    sm[tid] = mx; ss[tid] = s; __syncthreads();
    for (int off = 128; off > 0; off >>= 1) {
        if (tid < off) {
            const float m1 = sm[tid],       s1 = ss[tid];
            const float m2 = sm[tid + off], s2 = ss[tid + off];
            const float mm = fmaxf(m1, m2);
            sm[tid] = mm;
            ss[tid] = s1 * __expf(m1 - mm) + s2 * __expf(m2 - mm);
        }
        __syncthreads();
    }
    if (tid == 0) u[rid] = NORM_C - (sm[0] + __logf(ss[0]));
}

// ---------------- 3b) v[b,j] = norm - logsumexp_i(Z[b,i,j] + u[b,i]) ----------------
// grid (64, 32), block 256 = 32 cols x 8 row-groups
__global__ void sink_v_kernel(const float* __restrict__ Z, const float* __restrict__ u,
                              float* __restrict__ v) {
    const int b   = blockIdx.y;
    const int tid = threadIdx.x;
    const int tx  = tid & 31, ty = tid >> 5;
    const int j   = (blockIdx.x << 5) + tx;
    const float* zb = Z + (size_t)b * 256 * 2048 + j;
    const float* ub = u + b * 256;
    float mx = -3.0e38f, s = 0.f;
    for (int i = ty; i < 256; i += 8) {
        const float z = zb[(size_t)i * 2048] + ub[i];
        if (z > mx) { s = s * __expf(mx - z) + 1.f; mx = z; }
        else        { s += __expf(z - mx); }
    }
    __shared__ float sm[256], ss[256];
    sm[tid] = mx; ss[tid] = s; __syncthreads();
    for (int off = 4; off > 0; off >>= 1) {
        if (ty < off) {
            const int o = tid + (off << 5);
            const float m1 = sm[tid], s1 = ss[tid];
            const float m2 = sm[o],   s2 = ss[o];
            const float mm = fmaxf(m1, m2);
            sm[tid] = mm;
            ss[tid] = s1 * __expf(m1 - mm) + s2 * __expf(m2 - mm);
        }
        __syncthreads();
    }
    if (ty == 0) v[(size_t)b * 2048 + j] = NORM_C - (sm[tid] + __logf(ss[tid]));
}

// ---------------- 4) weights = exp(Z + u + v - norm): fp32 to out, bf16 for GEMM-2 ----
__global__ void weights_kernel(const float* __restrict__ Z, const float* __restrict__ u,
                               const float* __restrict__ v, float* __restrict__ wout,
                               unsigned short* __restrict__ wbf) {
    const int base = blockIdx.x * 2048 + threadIdx.x;
#pragma unroll
    for (int t = 0; t < 8; ++t) {
        const int idx = base + (t << 8);
        const int bm  = idx >> 11;
        const int j   = idx & 2047;
        const int b   = bm >> 8;
        const float wgt = __expf(Z[idx] + u[bm] + v[(b << 11) + j] - NORM_C);
        wout[idx] = wgt;
        wbf[idx]  = f2bf(wgt);
    }
}

// ---------------- 5) x_out[b,m,d] = sum_n W[b,m,n] * Xn[b,n,d] ----------------
// grid (6, 8, 32), block 256 = 8 waves (2 Mx x 4 Nx). Block tile: 32(M) x 128(D), K=2048.
// Double-buffered LDS B-tile (transposed for contiguous-K fragment reads); one barrier
// per K-step; A fragment double-buffered in registers with clamped addresses.
__global__ void out_wmma(const unsigned short* __restrict__ Wbf,
                         const unsigned short* __restrict__ xn,
                         float* __restrict__ xo) {
    const int tid  = threadIdx.x;
    const int wave = tid >> 5, lane = tid & 31;
    const int r16  = lane & 15, kh = lane >> 4;
    const int wm   = wave >> 2, wn = wave & 3;
    const int m0   = blockIdx.y << 5;
    const int d0   = blockIdx.x << 7;
    const int b    = blockIdx.z;

    __shared__ unsigned short Bt[2][128 * 40];   // [buf][col 0..127][k 0..31], pad stride 40

    const unsigned short* arow = Wbf + ((size_t)(b * 256 + m0 + wm * 16 + r16)) * 2048 + kh * 16;
    const unsigned short* xnb  = xn + (size_t)b * 2048 * 768;

    v8f acc0 = {0.f,0.f,0.f,0.f,0.f,0.f,0.f,0.f};
    v8f acc1 = acc0;

    // stage K-tile 0 into buffer 0
#pragma unroll
    for (int p = 0; p < 16; ++p) {
        const int idx = tid + (p << 8);
        const int rr = idx >> 7, cc = idx & 127;
        Bt[0][cc * 40 + rr] = xnb[(size_t)rr * 768 + d0 + cc];
    }
    v16bf A = load_a_frag(arow);

    for (int k = 0; k < 64; ++k) {
        __syncthreads();   // staging of Bt[k&1] complete; prior reads of Bt[(k+1)&1] done
        // stage next K-tile into the other buffer (clamped: restages last tile harmlessly)
        const int kn = (k < 63) ? (k + 1) : 63;
        const size_t kbase = (size_t)(kn << 5) * 768 + d0;
#pragma unroll
        for (int p = 0; p < 16; ++p) {
            const int idx = tid + (p << 8);
            const int rr = idx >> 7, cc = idx & 127;
            Bt[(k + 1) & 1][cc * 40 + rr] = xnb[kbase + (size_t)rr * 768 + cc];
        }
        v16bf An = load_a_frag(arow + (kn << 5));
        // consume current buffer
        const unsigned short* lb = &Bt[k & 1][(wn << 5) * 40];
        v16bf Bf0 = load_b_frag(lb + (size_t)r16 * 40 + kh * 16);
        v16bf Bf1 = load_b_frag(lb + (size_t)(16 + r16) * 40 + kh * 16);
        acc0 = __builtin_amdgcn_wmma_f32_16x16x32_bf16(false, A, false, Bf0, (short)0, acc0, false, false);
        acc1 = __builtin_amdgcn_wmma_f32_16x16x32_bf16(false, A, false, Bf1, (short)0, acc1, false, false);
        A = An;
    }

#pragma unroll
    for (int vv = 0; vv < 8; ++vv) {
        const int row = m0 + wm * 16 + vv + (kh << 3);
        float* op = xo + ((size_t)(b * 256 + row)) * 768 + d0 + (wn << 5) + r16;
        op[0]  = acc0[vv];
        op[16] = acc1[vv];
    }
}

// ---------------- host-side launch ----------------
extern "C" void kernel_launch(void* const* d_in, const int* in_sizes, int n_in,
                              void* d_out, int out_size, void* d_ws, size_t ws_size,
                              hipStream_t stream) {
    const float* x = (const float*)d_in[0];   // (32, 2048, 768) fp32
    const float* v = (const float*)d_in[1];   // (256, 768) fp32

    float* out   = (float*)d_out;
    float* out_x = out;                       // (32, 256, 768)
    float* out_w = out + 6291456;             // (32, 256, 2048)

    // workspace layout (bytes)
    char* ws = (char*)d_ws;
    unsigned short* xnb  = (unsigned short*)(ws + 0);            // 100,663,296 B  bf16 Xn
    unsigned short* wbq  = (unsigned short*)(ws + 100663296);    //     393,216 B  bf16 W
    float*          Z    = (float*)        (ws + 101056512);     //  67,108,864 B  fp32 Z
    float*          uu   = (float*)        (ws + 168165376);     //      32,768 B  u
    float*          vv   = (float*)        (ws + 168198144);     //     262,144 B  v
    unsigned short* Wbf  = (unsigned short*)(ws + 168460288);    //  33,554,432 B  bf16 weights

    // 1) normalize tokens and cluster centers -> bf16
    nrm_rows<<<65536, 256, 0, stream>>>(x, xnb);
    nrm_rows<<<256,   256, 0, stream>>>(v, wbq);

    // init dual potential v = 0 (u is overwritten before first use)
    hipMemsetAsync(vv, 0, (size_t)32 * 2048 * sizeof(float), stream);

    // 2) scores GEMM (WMMA bf16 -> fp32), scaled by 1/EPS
    score_wmma<<<dim3(4, 16, 32), 256, 0, stream>>>(xnb, wbq, Z);

    // 3) Sinkhorn iterations
    for (int it = 0; it < 5; ++it) {
        sink_u_kernel<<<8192, 256, 0, stream>>>(Z, vv, uu);
        sink_v_kernel<<<dim3(64, 32), 256, 0, stream>>>(Z, uu, vv);
    }

    // 4) weights: fp32 to d_out, bf16 copy for the second GEMM
    weights_kernel<<<8192, 256, 0, stream>>>(Z, uu, vv, out_w, Wbf);

    // 5) output GEMM (WMMA bf16 -> fp32)
    out_wmma<<<dim3(6, 8, 32), 256, 0, stream>>>(Wbf, xnb, out_x);
}
